// AttentionHead_15960098472459
// MI455X (gfx1250) — compile-verified, matrix operand order checked by reading, not run on hardware
//
#include <hip/hip_runtime.h>

// ---------------------------------------------------------------------------
// AttentionHead for MI455X (gfx1250, wave32, WMMA + Tensor Data Mover).
// q = x·Wq^T, k = pos·Wk^T, v = x·Wv^T  (bf16 operands, f32 accumulate)
// flash-attention with causal mask, scale = E^-0.5 = 1/64.
// ---------------------------------------------------------------------------

typedef __attribute__((ext_vector_type(16))) __bf16        v16bf;
typedef __attribute__((ext_vector_type(8)))  float         v8f;
typedef __attribute__((ext_vector_type(4)))  unsigned int  u32x4;
typedef __attribute__((ext_vector_type(4)))  int           i32x4;
typedef __attribute__((ext_vector_type(8)))  int           i32x8;

#define E_DIM   4096
#define H_DIM   128
#define T_DIM   2048
#define B_DIM   4
#define BT      (B_DIM * T_DIM)       // 8192 rows
#define KCHUNKS (E_DIM / 32)          // 128 K-chunks of 32
#define NTILES  (H_DIM / 16)          // 8 N-tiles of 16
// Per-matrix B-fragment array: [kc][ntile][lane(32)][16 bf16]
#define WFRAG_ELEMS (KCHUNKS * NTILES * 32 * 16)   // 524288 bf16 = 1 MB

#define USE_TDM __has_builtin(__builtin_amdgcn_tensor_load_to_lds)

// ---------------------------------------------------------------------------
// TDM: load one [16 rows x 32 cols] f32 tile of the [BT x E] tensor into LDS.
// D# per CDNA5 ISA:
//   group0: count=1, lds_addr, global_addr(57b), type=2
//   group1: data_size=4B(code 2), tensor_dim0=E, dim0_stride=E,
//           tensor_dim1=BT, tile_dim0=32, tile_dim1=16
// ---------------------------------------------------------------------------
#if USE_TDM
__device__ __forceinline__ void tdm_load_tile16x32(const float* gsrc,
                                                   float* ldsdst) {
  unsigned long long ga = (unsigned long long)(size_t)gsrc;
  unsigned int lds_lo = (unsigned int)(size_t)ldsdst;
  u32x4 g0 = { 1u,                                   // count=1, user mode
               lds_lo,                               // lds_addr [63:32]
               (unsigned int)ga,                     // global_addr [95:64]
               (unsigned int)((ga >> 32) & 0x1FFFFFFu) | 0x80000000u }; // +type=2
  i32x8 g1 = { 0x00020000,            // data_size=2 (4 bytes), mask=0
               0x10000000,            // tensor_dim0 low16 (4096) << 16
               0x20000000,            // dim0 hi16=0 | tensor_dim1 low16 (8192)<<16
               0x00200000,            // dim1 hi16=0 | tile_dim0 (32) << 16
               16,                    // tile_dim1=16, tile_dim2=0
               4096,                  // tensor_dim0_stride low32
               0, 0 };
  i32x4 gz = { 0, 0, 0, 0 };
#if __clang_major__ >= 23
  i32x8 gz8 = { 0, 0, 0, 0, 0, 0, 0, 0 };
  __builtin_amdgcn_tensor_load_to_lds(g0, g1, gz, gz, gz8, 0);
#else
  __builtin_amdgcn_tensor_load_to_lds(g0, g1, gz, gz, 0);
#endif
}
#endif

// ---------------------------------------------------------------------------
// Kernel 1: repack fp32 weights [H,E] into bf16 WMMA B-fragments.
// B-fragment element (lane, pos): n = lane%16, k_local = (lane<16?0:16)+pos
// so each lane's 16 bf16 are contiguous (one 32B load in the GEMM).
// ---------------------------------------------------------------------------
__global__ __launch_bounds__(256) void repack_w(
    const float* __restrict__ Wq, const float* __restrict__ Wk,
    const float* __restrict__ Wv, __bf16* __restrict__ out) {
  int idx  = blockIdx.x * 256 + threadIdx.x;       // [0, 3*WFRAG_ELEMS)
  int mat  = idx / WFRAG_ELEMS;
  int rem  = idx % WFRAG_ELEMS;
  int kc   = rem / (NTILES * 512);
  int rem2 = rem % (NTILES * 512);
  int nt   = rem2 / 512;
  int lane = (rem2 % 512) / 16;
  int pos  = rem2 % 16;
  int n      = lane & 15;
  int klocal = ((lane < 16) ? 0 : 16) + pos;
  const float* W = (mat == 0) ? Wq : ((mat == 1) ? Wk : Wv);
  out[idx] = (__bf16)W[(size_t)(nt * 16 + n) * E_DIM + kc * 32 + klocal];
}

// ---------------------------------------------------------------------------
// Kernel 2: fused QKV projection. One block = one 16-row M-tile.
// 8 waves: wave w owns output N-tile w (16 of 128 head dims) for q,k,v.
// x / pos_emb chunks DMA'd to LDS by the TDM (double-buffered, wave 0 issues,
// TENSORcnt + workgroup barrier publishes), bf16 cvt on the fragment build.
// ---------------------------------------------------------------------------
__global__ __launch_bounds__(256) void qkv_proj(
    const float* __restrict__ x, const float* __restrict__ pos,
    const __bf16* __restrict__ wfrag,
    __bf16* __restrict__ qb, __bf16* __restrict__ kb, __bf16* __restrict__ vb) {
  __shared__ float sx[2][512];   // [buf][16 rows x 32 cols] f32
  __shared__ float sp[2][512];

  const int tid  = threadIdx.x;
  const int wave = tid >> 5;
  const int lane = tid & 31;
  const int row0 = blockIdx.x * 16;
  const int m    = lane & 15;
  const int hsel = (lane < 16) ? 0 : 16;

  const __bf16* wq = wfrag;
  const __bf16* wk = wfrag + WFRAG_ELEMS;
  const __bf16* wv = wfrag + 2 * (size_t)WFRAG_ELEMS;

  v8f cq = {}, ck = {}, cv = {};

#if USE_TDM
  if (wave == 0) {   // uniform branch: TDM ignores EXEC, issue once per block
    tdm_load_tile16x32(x   + (size_t)row0 * E_DIM, &sx[0][0]);
    tdm_load_tile16x32(pos + (size_t)row0 * E_DIM, &sp[0][0]);
  }
#endif

  for (int kc = 0; kc < KCHUNKS; ++kc) {
    const int buf = kc & 1;
#if USE_TDM
    if (wave == 0) __builtin_amdgcn_s_wait_tensorcnt(0);
    __syncthreads();                       // publish tile kc to all waves
    if (wave == 0 && kc + 1 < KCHUNKS) {   // prefetch next chunk (other buf)
      size_t go = (size_t)row0 * E_DIM + (size_t)(kc + 1) * 32;
      tdm_load_tile16x32(x + go,   &sx[buf ^ 1][0]);
      tdm_load_tile16x32(pos + go, &sp[buf ^ 1][0]);
    }
#else
    // fallback: cooperative staged copy (f32, row-major [16][32])
    for (int t = tid; t < 512; t += 256) {
      size_t go = (size_t)(row0 + (t >> 5)) * E_DIM + (size_t)kc * 32 + (t & 31);
      sx[buf][t] = x[go];
      sp[buf][t] = pos[go];
      __builtin_prefetch(&x[go + 32], 0, 1);
      __builtin_prefetch(&pos[go + 32], 0, 1);
    }
    __syncthreads();
#endif

    // A-fragments: lane l -> row m = l%16, K = hsel + 0..15 (cvt f32->bf16)
    v16bf a_x, a_p;
    const float* bx = &sx[buf][m * 32 + hsel];
    const float* bp = &sp[buf][m * 32 + hsel];
    for (int i = 0; i < 16; ++i) {
      a_x[i] = (__bf16)bx[i];
      a_p[i] = (__bf16)bp[i];
    }

    size_t boff = ((size_t)(kc * NTILES + wave) * 32 + lane) * 16;
    v16bf bq = *(const v16bf*)&wq[boff];
    v16bf bk = *(const v16bf*)&wk[boff];
    v16bf bv = *(const v16bf*)&wv[boff];

    cq = __builtin_amdgcn_wmma_f32_16x16x32_bf16(false, a_x, false, bq,
                                                 (short)0, cq, false, false);
    ck = __builtin_amdgcn_wmma_f32_16x16x32_bf16(false, a_p, false, bk,
                                                 (short)0, ck, false, false);
    cv = __builtin_amdgcn_wmma_f32_16x16x32_bf16(false, a_x, false, bv,
                                                 (short)0, cv, false, false);
    __syncthreads();                       // done with buf before next DMA
  }

  // C/D layout: VGPR r, lane l -> row = r + (l<16?0:8), col = l%16
  const int half8 = (lane < 16) ? 0 : 8;
  for (int r = 0; r < 8; ++r) {
    size_t o = (size_t)(row0 + r + half8) * H_DIM + wave * 16 + m;
    qb[o] = (__bf16)cq[r];
    kb[o] = (__bf16)ck[r];
    vb[o] = (__bf16)cv[r];
  }
}

// ---------------------------------------------------------------------------
// Kernel 3: causal flash attention. One wave per 16-query tile; key tiles of
// 32. S in f32 (wmma bf16), online softmax in f32, P transposed via LDS,
// V tile staged to LDS with coalesced b128 loads, P·V with wmma.
// ---------------------------------------------------------------------------
__global__ __launch_bounds__(32) void attn(
    const __bf16* __restrict__ qb, const __bf16* __restrict__ kb,
    const __bf16* __restrict__ vb, float* __restrict__ out) {
  __shared__ __bf16 pl[16 * 32];     // P tile, row-major [16 q][32 key]
  __shared__ __bf16 vt[32 * 128];    // V tile, row-major [32 key][128 h]

  const int lane  = threadIdx.x & 31;
  const int q0    = blockIdx.x * 16;         // query tile within batch
  const int b     = blockIdx.y;
  const __bf16* qB = qb + (size_t)b * T_DIM * H_DIM;
  const __bf16* kB = kb + (size_t)b * T_DIM * H_DIM;
  const __bf16* vB = vb + (size_t)b * T_DIM * H_DIM;

  const int m     = lane & 15;               // A-frag row / B-frag col
  const int hsel  = (lane < 16) ? 0 : 16;    // K-range selector per lane half
  const int half8 = hsel >> 1;               // C-layout row offset (0 or 8)

  // q A-fragments for the 4 chunks of H=128
  v16bf qf[4];
  for (int hc = 0; hc < 4; ++hc)
    qf[hc] = *(const v16bf*)&qB[(size_t)(q0 + m) * H_DIM + hc * 32 + hsel];

  v8f acc[NTILES] = {};
  float mrow[8], lrow[8];
  for (int r = 0; r < 8; ++r) { mrow[r] = -__builtin_inff(); lrow[r] = 0.f; }

  const float sm  = 0.015625f;               // E^-0.5 = 1/64
  const int   nkt = (q0 + 47) >> 5;          // key tiles covering [0, q0+15]

  for (int kt = 0; kt < nkt; ++kt) {
    const int k0 = kt * 32;

    // ---- stage V tile [32 keys x 128 h] with coalesced 32B loads ----------
    for (int i = 0; i < 8; ++i) {
      int f = (i * 32 + lane) * 16;          // flat bf16 index in tile
      *(v16bf*)&vt[f] = *(const v16bf*)&vB[(size_t)k0 * H_DIM + f];
    }

    // ---- S = q · k^T over 4 H-chunks, two 16-key halves -------------------
    v8f s0 = {}, s1 = {};
    for (int hc = 0; hc < 4; ++hc) {
      v16bf b0 = *(const v16bf*)&kB[(size_t)(k0 + m) * H_DIM + hc * 32 + hsel];
      v16bf b1 = *(const v16bf*)&kB[(size_t)(k0 + 16 + m) * H_DIM + hc * 32 + hsel];
      s0 = __builtin_amdgcn_wmma_f32_16x16x32_bf16(false, qf[hc], false, b0,
                                                   (short)0, s0, false, false);
      s1 = __builtin_amdgcn_wmma_f32_16x16x32_bf16(false, qf[hc], false, b1,
                                                   (short)0, s1, false, false);
    }

    // ---- scale + causal mask (lane's column n = m) ------------------------
    for (int r = 0; r < 8; ++r) {
      int qrow = q0 + r + half8;
      float v0 = s0[r] * sm, v1 = s1[r] * sm;
      s0[r] = (k0 + m      <= qrow) ? v0 : -__builtin_inff();
      s1[r] = (k0 + 16 + m <= qrow) ? v1 : -__builtin_inff();
    }

    // ---- online softmax (reductions stay inside 16-lane row-halves) -------
    for (int r = 0; r < 8; ++r) {
      float cmax = fmaxf(s0[r], s1[r]);
      for (int off = 8; off >= 1; off >>= 1)
        cmax = fmaxf(cmax, __shfl_xor(cmax, off, 32));
      float mnew  = fmaxf(mrow[r], cmax);
      float scale = __expf(mrow[r] - mnew);
      float p0 = __expf(s0[r] - mnew);
      float p1 = __expf(s1[r] - mnew);
      float csum = p0 + p1;
      for (int off = 8; off >= 1; off >>= 1)
        csum += __shfl_xor(csum, off, 32);
      lrow[r] = lrow[r] * scale + csum;
      mrow[r] = mnew;
      s0[r] = p0; s1[r] = p1;
      for (int nt = 0; nt < NTILES; ++nt) acc[nt][r] *= scale;
    }

    // ---- transpose P (C-layout) into an A-fragment via LDS ----------------
    __syncthreads();   // previous pf reads done before overwrite
    for (int r = 0; r < 8; ++r) {
      pl[(r + half8) * 32 + m]      = (__bf16)s0[r];
      pl[(r + half8) * 32 + 16 + m] = (__bf16)s1[r];
    }
    __syncthreads();
    v16bf pf = *(const v16bf*)&pl[m * 32 + hsel];

    // ---- acc += P · V (V B-fragments gathered from LDS) -------------------
    for (int nt = 0; nt < NTILES; ++nt) {
      v16bf vf;
      for (int i = 0; i < 16; ++i)
        vf[i] = vt[(hsel + i) * H_DIM + nt * 16 + m];
      acc[nt] = __builtin_amdgcn_wmma_f32_16x16x32_bf16(false, pf, false, vf,
                                                        (short)0, acc[nt],
                                                        false, false);
    }
    __syncthreads();   // done with vt before next tile's staging
  }

  // ---- normalize + write out (f32) ----------------------------------------
  for (int nt = 0; nt < NTILES; ++nt)
    for (int r = 0; r < 8; ++r) {
      size_t o = ((size_t)b * T_DIM + q0 + r + half8) * H_DIM + nt * 16 + m;
      out[o] = acc[nt][r] / lrow[r];
    }
}

// ---------------------------------------------------------------------------
extern "C" void kernel_launch(void* const* d_in, const int* in_sizes, int n_in,
                              void* d_out, int out_size, void* d_ws,
                              size_t ws_size, hipStream_t stream) {
  const float* x   = (const float*)d_in[0];
  const float* pos = (const float*)d_in[1];
  const float* Wq  = (const float*)d_in[2];
  const float* Wk  = (const float*)d_in[3];
  const float* Wv  = (const float*)d_in[4];
  float* out = (float*)d_out;

  // workspace: [3 x weight frags (1MB ea)] [q|k|v bf16 (2MB ea)] = 9 MB
  __bf16* wfrag = (__bf16*)d_ws;
  __bf16* qb = wfrag + 3 * (size_t)WFRAG_ELEMS;
  __bf16* kb = qb + (size_t)BT * H_DIM;
  __bf16* vb = kb + (size_t)BT * H_DIM;

  repack_w<<<(3 * WFRAG_ELEMS) / 256, 256, 0, stream>>>(Wq, Wk, Wv, wfrag);
  qkv_proj<<<BT / 16, 256, 0, stream>>>(x, pos, wfrag, qb, kb, vb);
  attn<<<dim3(T_DIM / 16, B_DIM), 32, 0, stream>>>(qb, kb, vb, out);
}
